// Encoder_85332410237214
// MI455X (gfx1250) — matrix operand order repaired and checked by reading, not compiled, not run on hardware
//
#include <hip/hip_runtime.h>

typedef __attribute__((ext_vector_type(16))) _Float16 v16h;
typedef __attribute__((ext_vector_type(8)))  _Float16 v8h;
typedef __attribute__((ext_vector_type(8)))  float    v8f;
typedef __attribute__((ext_vector_type(4)))  unsigned int v4u;
typedef __attribute__((ext_vector_type(4)))  int      v4i;

#define NPTS   256          // points per block (= threads per block)
#define PITCH  136          // halves per activation row (128 + 8 pad; 68-dword stride, conflict-free)
#define W0P    104          // halves per w0 row (96 + 8 pad; 52-dword stride, conflict-free)
#define WP     136          // halves per w1/w2/w3 row

#define BUF0_OFF 0
#define BUF1_OFF (NPTS * PITCH)                 // 34816 halves
#define WBASE    (2 * NPTS * PITCH)             // 69632 halves
#define W0_OFF   (WBASE)
#define W1_OFF   (W0_OFF + 128 * W0P)           // +13312
#define W2_OFF   (W1_OFF + 128 * WP)            // +17408
#define W3_OFF   (W2_OFF + 128 * WP)            // +17408
#define WS_HALVES (128 * W0P + 2 * 128 * WP + 16 * WP)   // 50304 halves (100608 B)
#define LDS_BYTES ((WBASE + WS_HALVES) * 2)     // 239872 B < 320 KB

// Async global->LDS path (CDNA5): bypasses VGPRs, tracked by ASYNCcnt.
#if defined(__has_builtin)
#  if __has_builtin(__builtin_amdgcn_global_load_async_to_lds_b128) && \
      __has_builtin(__builtin_amdgcn_s_wait_asynccnt)
#    define USE_ASYNC_LDS 1
#  endif
#endif
#ifndef USE_ASYNC_LDS
#  define USE_ASYNC_LDS 0
#endif

#define AS_GLOBAL __attribute__((address_space(1)))
#define AS_LOCAL  __attribute__((address_space(3)))

// ---------------- prep: f32 weights -> padded/pitched f16 image in ws ----------------
__global__ void prep_weights(const float* __restrict__ w0, const float* __restrict__ w1,
                             const float* __restrict__ w2, const float* __restrict__ w3,
                             _Float16* __restrict__ ws)
{
    int h = blockIdx.x * blockDim.x + threadIdx.x;
    if (h >= WS_HALVES) return;
    float v = 0.0f;
    if (h < 128 * W0P) {
        int r = h / W0P, c = h % W0P;
        if (c < 96) v = w0[r * 96 + c];
    } else if (h < 128 * W0P + 128 * WP) {
        int j = h - 128 * W0P; int r = j / WP, c = j % WP;
        if (c < 128) v = w1[r * 128 + c];
    } else if (h < 128 * W0P + 2 * 128 * WP) {
        int j = h - (128 * W0P + 128 * WP); int r = j / WP, c = j % WP;
        if (c < 128) v = w2[r * 128 + c];
    } else {
        int j = h - (128 * W0P + 2 * 128 * WP); int r = j / WP, c = j % WP;
        if (r < 3 && c < 128) v = w3[r * 128 + c];
    }
    ws[h] = (_Float16)v;
}

// ---------------- fragment loaders (LDS) ----------------
// A (16x32 f16): lane<16 holds row M=lane, K {0..7, 16..23}; lane>=16 row M=lane-16, K {8..15, 24..31}
__device__ __forceinline__ v16h load_a(const _Float16* buf, int m0, int k0, int lane)
{
    const _Float16* p = buf + (m0 + (lane & 15)) * PITCH + k0 + ((lane >> 4) * 8);
    v8h lo = *(const v8h*)(p);
    v8h hi = *(const v8h*)(p + 16);
    return __builtin_shufflevector(lo, hi, 0, 1, 2, 3, 4, 5, 6, 7, 8, 9, 10, 11, 12, 13, 14, 15);
}

// B (32x16 f16): lane<16 holds col N=lane, K {0..7, 16..23}; lane>=16 col N=lane-16, K {8..15, 24..31}
__device__ __forceinline__ v16h load_b(const _Float16* W, int wpitch, int n0, int k0, int lane)
{
    const _Float16* p = W + (n0 + (lane & 15)) * wpitch + k0 + ((lane >> 4) * 8);
    v8h lo = *(const v8h*)(p);
    v8h hi = *(const v8h*)(p + 16);
    return __builtin_shufflevector(lo, hi, 0, 1, 2, 3, 4, 5, 6, 7, 8, 9, 10, 11, 12, 13, 14, 15);
}

// ---------------- one hidden layer: src(LDS f16) x W(LDS f16) + b -> relu -> dst(LDS f16) ----------------
template <int KCH, int NT, int WPITCH>
__device__ __forceinline__ void mlp_layer(const _Float16* src, _Float16* dst,
                                          const _Float16* W, const float* __restrict__ bias,
                                          int lane, int wave)
{
    v8f acc[2][NT];
#pragma unroll
    for (int t = 0; t < 2; ++t)
#pragma unroll
        for (int n = 0; n < NT; ++n) acc[t][n] = (v8f){};

#pragma unroll
    for (int k = 0; k < KCH; ++k) {
        v16h a0 = load_a(src, wave * 32,      k * 32, lane);
        v16h a1 = load_a(src, wave * 32 + 16, k * 32, lane);
#pragma unroll
        for (int n = 0; n < NT; ++n) {
            v16h b = load_b(W, WPITCH, n * 16, k * 32, lane);
            acc[0][n] = __builtin_amdgcn_wmma_f32_16x16x32_f16(false, a0, false, b, (short)0, acc[0][n], false, false);
            acc[1][n] = __builtin_amdgcn_wmma_f32_16x16x32_f16(false, a1, false, b, (short)0, acc[1][n], false, false);
        }
    }
    // epilogue: D layout = lane holds col N=lane&15; VGPR r holds row r + (lane>=16 ? 8 : 0)
#pragma unroll
    for (int n = 0; n < NT; ++n) {
        float bz = bias[n * 16 + (lane & 15)];
        int col = n * 16 + (lane & 15);
#pragma unroll
        for (int t = 0; t < 2; ++t) {
            int row0 = wave * 32 + t * 16 + ((lane >> 4) * 8);
#pragma unroll
            for (int r = 0; r < 8; ++r) {
                float v = fmaxf(acc[t][n][r] + bz, 0.0f);
                dst[(row0 + r) * PITCH + col] = (_Float16)v;
            }
        }
    }
}

// ---------------- output layer: 128 -> 3 (one N-tile, no relu, store f32 to global) ----------------
template <int KCH>
__device__ __forceinline__ void mlp_out(const _Float16* src, const _Float16* W,
                                        const float* __restrict__ bias,
                                        float* __restrict__ out, int base, int lane, int wave)
{
    v8f acc[2];
    acc[0] = (v8f){}; acc[1] = (v8f){};
#pragma unroll
    for (int k = 0; k < KCH; ++k) {
        v16h a0 = load_a(src, wave * 32,      k * 32, lane);
        v16h a1 = load_a(src, wave * 32 + 16, k * 32, lane);
        v16h b  = load_b(W, WP, 0, k * 32, lane);
        acc[0] = __builtin_amdgcn_wmma_f32_16x16x32_f16(false, a0, false, b, (short)0, acc[0], false, false);
        acc[1] = __builtin_amdgcn_wmma_f32_16x16x32_f16(false, a1, false, b, (short)0, acc[1], false, false);
    }
    int c = lane & 15;
    if (c < 3) {
        float bz = bias[c];
#pragma unroll
        for (int t = 0; t < 2; ++t) {
            int row0 = base + wave * 32 + t * 16 + ((lane >> 4) * 8);
#pragma unroll
            for (int r = 0; r < 8; ++r)
                out[(row0 + r) * 3 + c] = acc[t][r] + bz;
        }
    }
}

// ---------------- bilinear sample of one LOD (16 channels) ----------------
__device__ __forceinline__ void sample_lod(const float* __restrict__ fm, int s, float g,
                                           float x, float y, _Float16* dstRow)
{
    float fs = (float)(s - 1);
    float fx = fminf(fmaxf((x + 1.0f) * 0.5f * fs, 0.0f), fs);
    float fy = fminf(fmaxf((y + 1.0f) * 0.5f * fs, 0.0f), fs);
    float x0f = floorf(fx), y0f = floorf(fy);
    float wx = fx - x0f, wy = fy - y0f;
    int x0 = (int)x0f, y0 = (int)y0f;
    int x1 = (x0 + 1 < s - 1) ? x0 + 1 : s - 1;
    int y1 = (y0 + 1 < s - 1) ? y0 + 1 : s - 1;
    float w00 = (1.0f - wx) * (1.0f - wy);
    float w10 = wx * (1.0f - wy);
    float w01 = (1.0f - wx) * wy;
    float w11 = wx * wy;
    int ss = s * s;
    int b00 = y0 * s + x0, b10 = y0 * s + x1, b01 = y1 * s + x0, b11 = y1 * s + x1;
#pragma unroll
    for (int c = 0; c < 16; ++c) {
        int o = c * ss;
        float v = fm[o + b00] * w00 + fm[o + b10] * w10 + fm[o + b01] * w01 + fm[o + b11] * w11;
        dstRow[c] = (_Float16)(v * g);
    }
}

// ---------------- fused encoder kernel ----------------
__global__ void __launch_bounds__(NPTS)
encoder_kernel(const float* __restrict__ xc,
               const float* __restrict__ fm0, const float* __restrict__ fm1,
               const float* __restrict__ fm2, const float* __restrict__ fm3,
               const float* __restrict__ fm4, const float* __restrict__ fm5,
               const float* __restrict__ lg,
               const float* __restrict__ b0, const float* __restrict__ b1,
               const float* __restrict__ b2, const float* __restrict__ b3,
               const _Float16* __restrict__ ws,
               float* __restrict__ out)
{
    extern __shared__ _Float16 smem[];
    _Float16* buf0 = smem + BUF0_OFF;
    _Float16* buf1 = smem + BUF1_OFF;

    int tid = threadIdx.x;
    __builtin_prefetch((const void*)ws, 0, 1);   // global_prefetch: warm weight image into caches

    // cooperative global->LDS copy of the padded weight image (exactly 6288 x 16B)
#if USE_ASYNC_LDS
    {
        AS_GLOBAL v4i* s = (AS_GLOBAL v4i*)ws;
        AS_LOCAL  v4i* d = (AS_LOCAL  v4i*)(smem + WBASE);
        for (int i = tid; i < (WS_HALVES / 8); i += NPTS)
            __builtin_amdgcn_global_load_async_to_lds_b128(s + i, d + i, 0, 0);
    }
#else
    {
        const v4u* s = (const v4u*)ws;
        v4u*       d = (v4u*)(smem + WBASE);
        for (int i = tid; i < (WS_HALVES / 8); i += NPTS) d[i] = s[i];
    }
#endif

    // feature sampling: one point per thread into buf0 (layer-0 A operand)
    int base = blockIdx.x * NPTS;
    int p = base + tid;
    float x = xc[2 * p + 0];
    float y = xc[2 * p + 1];
    _Float16* row = buf0 + tid * PITCH;
    sample_lod(fm0,  16, exp2f(lg[0]), x, y, row + 0);
    sample_lod(fm1,  32, exp2f(lg[1]), x, y, row + 16);
    sample_lod(fm2,  64, exp2f(lg[2]), x, y, row + 32);
    sample_lod(fm3, 128, exp2f(lg[3]), x, y, row + 48);
    sample_lod(fm4, 256, exp2f(lg[4]), x, y, row + 64);
    sample_lod(fm5, 512, exp2f(lg[5]), x, y, row + 80);

#if USE_ASYNC_LDS
    __builtin_amdgcn_s_wait_asynccnt(0);   // async weight-image writes visible before barrier
#endif
    __syncthreads();

    int lane = tid & 31;
    int wave = tid >> 5;

    mlp_layer<3, 8, W0P>(buf0, buf1, smem + W0_OFF, b0, lane, wave);   // 96 -> 128
    __syncthreads();
    mlp_layer<4, 8, WP >(buf1, buf0, smem + W1_OFF, b1, lane, wave);   // 128 -> 128
    __syncthreads();
    mlp_layer<4, 8, WP >(buf0, buf1, smem + W2_OFF, b2, lane, wave);   // 128 -> 128
    __syncthreads();
    mlp_out<4>(buf1, smem + W3_OFF, b3, out, base, lane, wave);        // 128 -> 3
}

extern "C" void kernel_launch(void* const* d_in, const int* in_sizes, int n_in,
                              void* d_out, int out_size, void* d_ws, size_t ws_size,
                              hipStream_t stream)
{
    (void)in_sizes; (void)n_in; (void)out_size; (void)ws_size;
    const float* xc  = (const float*)d_in[0];
    const float* fm0 = (const float*)d_in[1];
    const float* fm1 = (const float*)d_in[2];
    const float* fm2 = (const float*)d_in[3];
    const float* fm3 = (const float*)d_in[4];
    const float* fm4 = (const float*)d_in[5];
    const float* fm5 = (const float*)d_in[6];
    const float* lg  = (const float*)d_in[7];
    const float* w0  = (const float*)d_in[8];
    const float* b0  = (const float*)d_in[9];
    const float* w1  = (const float*)d_in[10];
    const float* b1  = (const float*)d_in[11];
    const float* w2  = (const float*)d_in[12];
    const float* b2  = (const float*)d_in[13];
    const float* w3  = (const float*)d_in[14];
    const float* b3  = (const float*)d_in[15];
    _Float16* ws = (_Float16*)d_ws;
    float* out = (float*)d_out;

    prep_weights<<<(WS_HALVES + 255) / 256, 256, 0, stream>>>(w0, w1, w2, w3, ws);

    const int npoints = 1024 * 1024;
    encoder_kernel<<<npoints / NPTS, NPTS, LDS_BYTES, stream>>>(
        xc, fm0, fm1, fm2, fm3, fm4, fm5, lg, b0, b1, b2, b3,
        (const _Float16*)ws, out);
}
